// TernaryLinear_31851477467321
// MI455X (gfx1250) — matrix operand order repaired and checked by reading, not compiled, not run on hardware
//
#include <hip/hip_runtime.h>

typedef _Float16 v16h __attribute__((ext_vector_type(16)));
typedef _Float16 v8h  __attribute__((ext_vector_type(8)));
typedef float    v8f  __attribute__((ext_vector_type(8)));
typedef float    v4f  __attribute__((ext_vector_type(4)));

#define B_DIM   32
#define IN_DIM  4096
#define OUT_DIM 11008
#define NW      45088768   // OUT_DIM * IN_DIM
#define RED_BLOCKS  1024
#define RED_THREADS 256
#define KSPLIT  2
#define KHALF   (IN_DIM / KSPLIT)          // 2048
#define NBLOCKS (OUT_DIM / 128)            // 86 n-blocks (128 cols each)
#define OUT_ELEMS (B_DIM * OUT_DIM)        // 352256

// ---------------------------------------------------------------------------
// Workspace layout (d_ws):
//   byte 0       : float delta
//   byte 256     : float partials[RED_BLOCKS]            (4 KB)
//   byte 8192    : _Float16 xh[B_DIM * IN_DIM]           (256 KB)
//   byte 270336  : float part[KSPLIT][B_DIM][OUT_DIM]    (2.75 MB)
// ---------------------------------------------------------------------------
#define XH_OFF   8192
#define PART_OFF 270336

// Pass 1: per-block partial sums of |W|. Regular (RT) loads on purpose:
// W (180 MB) fits in the 192 MB L2, so this pass warms L2 for the GEMM pass.
__global__ void __launch_bounds__(RED_THREADS)
ternlin_reduce_abs(const float* __restrict__ w, float* __restrict__ partials) {
    __shared__ float sm[RED_THREADS];
    const int tid = threadIdx.x;
    size_t gid    = (size_t)blockIdx.x * blockDim.x + tid;
    size_t stride = (size_t)gridDim.x * blockDim.x;
    const v4f* w4 = (const v4f*)w;
    const size_t n4 = NW / 4;
    float s = 0.0f;
    for (size_t i = gid; i < n4; i += stride) {
        v4f v = w4[i];
        s += fabsf(v.x) + fabsf(v.y) + fabsf(v.z) + fabsf(v.w);
    }
    sm[tid] = s;
    __syncthreads();
    for (int off = RED_THREADS / 2; off > 0; off >>= 1) {
        if (tid < off) sm[tid] += sm[tid + off];
        __syncthreads();
    }
    if (tid == 0) partials[blockIdx.x] = sm[0];
}

// Pass 2: single block sums 1024 partials in fixed order -> deterministic delta.
__global__ void __launch_bounds__(RED_THREADS)
ternlin_finalize_delta(const float* __restrict__ partials, float* __restrict__ delta_out) {
    __shared__ float sm[RED_THREADS];
    const int tid = threadIdx.x;
    float s = 0.0f;
    for (int i = tid; i < RED_BLOCKS; i += RED_THREADS) s += partials[i];
    sm[tid] = s;
    __syncthreads();
    for (int off = RED_THREADS / 2; off > 0; off >>= 1) {
        if (tid < off) sm[tid] += sm[tid + off];
        __syncthreads();
    }
    if (tid == 0) delta_out[0] = 0.7f * (sm[0] / (float)NW);
}

// Convert activations x (f32) -> f16 once (ternary weights are exact in f16).
__global__ void __launch_bounds__(256)
ternlin_convert_x(const float* __restrict__ x, _Float16* __restrict__ xh) {
    int i = blockIdx.x * blockDim.x + threadIdx.x;
    if (i < B_DIM * IN_DIM) xh[i] = (_Float16)x[i];
}

__device__ __forceinline__ _Float16 tern_h(float w, float d) {
    return (_Float16)((w > d) ? 1.0f : ((w < -d) ? -1.0f : 0.0f));
}

// GEMM: each wave owns M=32 x N=16 over one K-half (2048) => each W element is
// loaded and ternarized exactly ONCE; one B fragment feeds two WMMAs.
// Block = 256 thr = 8 waves => N=128 per block. Grid = 86 n-blocks x 2 k-halves
// = 172 blocks / 1376 waves. W reads are NT (last use; L2-resident from pass 1).
__global__ void __launch_bounds__(256)
ternlin_wmma_gemm(const _Float16* __restrict__ xh,
                  const float*    __restrict__ w,
                  const float*    __restrict__ delta_p,
                  float*          __restrict__ part) {
    const int tid  = threadIdx.x;
    const int wave = tid >> 5;
    const int lane = tid & 31;
    const int hl   = lane & 15;     // lane column (B/C/D) / lane row (A)
    const int hi   = lane >> 4;     // lane-group select

    const int nb = blockIdx.x % NBLOCKS;     // n-block
    const int kh = blockIdx.x / NBLOCKS;     // k-half: 0 or 1
    const int n  = nb * 128 + wave * 16 + hl;
    const int k0 = kh * KHALF;

    const float delta = delta_p[0];

    const _Float16* arow0 = xh + (size_t)hl * IN_DIM;         // x rows 0..15
    const _Float16* arow1 = xh + (size_t)(16 + hl) * IN_DIM;  // x rows 16..31
    const float*    wrow  = w  + (size_t)n * IN_DIM;

    v8f acc0 = {};   // M rows 0..15
    v8f acc1 = {};   // M rows 16..31

    #pragma unroll 2
    for (int kb = k0; kb < k0 + KHALF; kb += 32) {
        // ---- A fragments (16x32 f16): lanes 0-15: K=kb..+15;
        //                               lanes 16-31: K=kb+16..+31.
        const int kf = kb + hi * 16;
        const int ks = kb + hi * 16 + 8;
        v8h a0l = *(const v8h*)(arow0 + kf);
        v8h a0h = *(const v8h*)(arow0 + ks);
        v8h a1l = *(const v8h*)(arow1 + kf);
        v8h a1h = *(const v8h*)(arow1 + ks);
        v16h afrag0 = __builtin_shufflevector(a0l, a0h,
                          0,1,2,3,4,5,6,7,8,9,10,11,12,13,14,15);
        v16h afrag1 = __builtin_shufflevector(a1l, a1h,
                          0,1,2,3,4,5,6,7,8,9,10,11,12,13,14,15);

        // ---- B fragment (32x16 f16): lane col = hl, K = kb + hi*16 + 0..15,
        //      contiguous along IN in W => 4x v4f NT loads + ternarize (once!).
        const float* wp = wrow + kb + hi * 16;
        v4f w0 = __builtin_nontemporal_load((const v4f*)(wp + 0));
        v4f w1 = __builtin_nontemporal_load((const v4f*)(wp + 4));
        v4f w2 = __builtin_nontemporal_load((const v4f*)(wp + 8));
        v4f w3 = __builtin_nontemporal_load((const v4f*)(wp + 12));
        v16h bfrag;
        bfrag[0]  = tern_h(w0.x, delta); bfrag[1]  = tern_h(w0.y, delta);
        bfrag[2]  = tern_h(w0.z, delta); bfrag[3]  = tern_h(w0.w, delta);
        bfrag[4]  = tern_h(w1.x, delta); bfrag[5]  = tern_h(w1.y, delta);
        bfrag[6]  = tern_h(w1.z, delta); bfrag[7]  = tern_h(w1.w, delta);
        bfrag[8]  = tern_h(w2.x, delta); bfrag[9]  = tern_h(w2.y, delta);
        bfrag[10] = tern_h(w2.z, delta); bfrag[11] = tern_h(w2.w, delta);
        bfrag[12] = tern_h(w3.x, delta); bfrag[13] = tern_h(w3.y, delta);
        bfrag[14] = tern_h(w3.z, delta); bfrag[15] = tern_h(w3.w, delta);

        // Two D = A x B + C sharing one B fragment.
        acc0 = __builtin_amdgcn_wmma_f32_16x16x32_f16(
                   false, afrag0, false, bfrag, (short)0, acc0, false, false);
        acc1 = __builtin_amdgcn_wmma_f32_16x16x32_f16(
                   false, afrag1, false, bfrag, (short)0, acc1, false, false);
    }

    // ---- Store partials: D VGPR r -> M = hi*8 + r (tile0) / 16 + hi*8 + r (tile1).
    float* pbase = part + (size_t)kh * OUT_ELEMS;
    #pragma unroll
    for (int r = 0; r < 8; ++r) {
        const int m0 = hi * 8 + r;
        pbase[(size_t)m0 * OUT_DIM + n]        = acc0[r];
        pbase[(size_t)(16 + m0) * OUT_DIM + n] = acc1[r];
    }
}

// Combine the two K-half partials in fixed order; apply alpha and bias.
__global__ void __launch_bounds__(256)
ternlin_combine(const float* __restrict__ part,
                const float* __restrict__ alpha_p,
                const float* __restrict__ bias,
                float* __restrict__ out) {
    int i = blockIdx.x * blockDim.x + threadIdx.x;
    if (i < OUT_ELEMS) {
        int nn = i % OUT_DIM;
        out[i] = alpha_p[0] * (part[i] + part[i + OUT_ELEMS]) + bias[nn];
    }
}

extern "C" void kernel_launch(void* const* d_in, const int* in_sizes, int n_in,
                              void* d_out, int out_size, void* d_ws, size_t ws_size,
                              hipStream_t stream) {
    const float* x     = (const float*)d_in[0];   // [32, 4096]
    const float* w     = (const float*)d_in[1];   // [11008, 4096]
    const float* alpha = (const float*)d_in[2];   // [1]
    const float* bias  = (const float*)d_in[3];   // [11008]
    float*       out   = (float*)d_out;           // [32, 11008]

    char* ws        = (char*)d_ws;
    float* delta    = (float*)(ws + 0);
    float* partials = (float*)(ws + 256);
    _Float16* xh    = (_Float16*)(ws + XH_OFF);
    float* part     = (float*)(ws + PART_OFF);

    ternlin_reduce_abs<<<RED_BLOCKS, RED_THREADS, 0, stream>>>(w, partials);
    ternlin_finalize_delta<<<1, RED_THREADS, 0, stream>>>(partials, delta);
    ternlin_convert_x<<<(B_DIM * IN_DIM + 255) / 256, 256, 0, stream>>>(x, xh);
    ternlin_wmma_gemm<<<NBLOCKS * KSPLIT, 256, 0, stream>>>(xh, w, delta, part);
    ternlin_combine<<<(OUT_ELEMS + 255) / 256, 256, 0, stream>>>(part, alpha, bias, out);
}
